// ConformerLayer_9663676416110
// MI455X (gfx1250) — compile-verified, matrix operand order checked by reading, not run on hardware
//
#include <hip/hip_runtime.h>

// ---------------- types ----------------
typedef __bf16 bf16;
typedef __attribute__((ext_vector_type(16))) __bf16 v16bf;
typedef __attribute__((ext_vector_type(8)))  __bf16 bf16x8;
typedef __attribute__((ext_vector_type(8)))  float  v8f;
typedef int v4i_ __attribute__((vector_size(4 * sizeof(int))));
typedef __attribute__((address_space(1))) v4i_* gv4i_p;   // global 128-bit ptr
typedef __attribute__((address_space(3))) v4i_* lv4i_p;   // LDS 128-bit ptr

#define Dm   512
#define Fm   2048
#define Hh   8
#define KK   31
#define Tt   1024
#define Bb   16
#define HDd  64
#define NROW (Tt * Bb)          // 16384
#define EPSf 1e-5f

// CDNA5 async global->LDS DMA copy (ASYNCcnt-tracked)
#define ASYNC_LDS_B128(gp, lp)                                              \
  __builtin_amdgcn_global_load_async_to_lds_b128(                           \
      (gv4i_p)(const void*)(gp), (lv4i_p)(void*)(lp), 0, 0)

// ---------------- helpers ----------------
__device__ __forceinline__ float b2f(bf16 b) {
  unsigned short s = __builtin_bit_cast(unsigned short, b);
  return __uint_as_float(((unsigned)s) << 16);
}
__device__ __forceinline__ bf16 f2b(float f) {
  unsigned u = __float_as_uint(f);
  unsigned r = (u + 0x7FFFu + ((u >> 16) & 1u)) >> 16;
  unsigned short s = (unsigned short)r;
  return __builtin_bit_cast(bf16, s);
}
__device__ __forceinline__ float sigmoidf_(float x) { return 1.f / (1.f + __expf(-x)); }

__device__ __forceinline__ v16bf make_frag(bf16x8 lo, bf16x8 hi) {
  union { v16bf v; bf16x8 h[2]; } u;
  u.h[0] = lo; u.h[1] = hi;
  return u.v;
}
__device__ __forceinline__ float redmax16(float v) {
  v = fmaxf(v, __shfl_xor(v, 1, 16));
  v = fmaxf(v, __shfl_xor(v, 2, 16));
  v = fmaxf(v, __shfl_xor(v, 4, 16));
  v = fmaxf(v, __shfl_xor(v, 8, 16));
  return v;
}
__device__ __forceinline__ float redsum16(float v) {
  v += __shfl_xor(v, 1, 16);
  v += __shfl_xor(v, 2, 16);
  v += __shfl_xor(v, 4, 16);
  v += __shfl_xor(v, 8, 16);
  return v;
}
__device__ __forceinline__ float redsum32(float v) {
  v += __shfl_xor(v, 1);
  v += __shfl_xor(v, 2);
  v += __shfl_xor(v, 4);
  v += __shfl_xor(v, 8);
  v += __shfl_xor(v, 16);
  return v;
}

// ---------------- fused transpose + fp32->bf16 weight convert ----------------
// w: [Kd][N] f32 row-major -> wt: [N][Kd] bf16 row-major
__global__ __launch_bounds__(256) void wtr_kernel(const float* __restrict__ w,
                                                  bf16* __restrict__ wt,
                                                  int Kd, int N) {
  __shared__ float tile[32][33];
  const int k0 = blockIdx.y * 32, n0 = blockIdx.x * 32;
  const int nl = threadIdx.x & 31, r = threadIdx.x >> 5;
#pragma unroll
  for (int j = 0; j < 4; ++j) {
    int kl = r + j * 8;
    tile[kl][nl] = w[(size_t)(k0 + kl) * N + n0 + nl];
  }
  __syncthreads();
#pragma unroll
  for (int j = 0; j < 4; ++j) {
    int nl2 = r + j * 8;
    wt[(size_t)(n0 + nl2) * Kd + k0 + nl] = f2b(tile[nl][nl2]);
  }
}

// ---------------- LayerNorm over last dim (512) ----------------
template <int OUTMODE>   // 0: bf16 out, 1: f32 out
__global__ __launch_bounds__(256) void ln_kernel(const float* __restrict__ x,
                                                 const float* __restrict__ g,
                                                 const float* __restrict__ bta,
                                                 bf16* __restrict__ outB,
                                                 float* __restrict__ outF) {
  const int w = threadIdx.x >> 5, lane = threadIdx.x & 31;
  const int row = blockIdx.x * 8 + w;
  const float* xr = x + (size_t)row * Dm;
  float v[16];
  float s = 0.f;
#pragma unroll
  for (int i = 0; i < 16; ++i) { v[i] = xr[lane + 32 * i]; s += v[i]; }
  s = redsum32(s);
  const float mean = s * (1.f / Dm);
  float q = 0.f;
#pragma unroll
  for (int i = 0; i < 16; ++i) { float c = v[i] - mean; q += c * c; }
  q = redsum32(q);
  const float rstd = rsqrtf(q * (1.f / Dm) + EPSf);
#pragma unroll
  for (int i = 0; i < 16; ++i) {
    int c = lane + 32 * i;
    float o = (v[i] - mean) * rstd * g[c] + bta[c];
    if (OUTMODE == 0) outB[(size_t)row * Dm + c] = f2b(o);
    else              outF[(size_t)row * Dm + c] = o;
  }
}

// ---------------- WMMA bf16 GEMM: C = A[MxK] * Bt[NxK]^T (+epilogue) ----------------
// Async global->LDS staging, double-buffered, BK=64.
// MODE 0: f32 out = acc+bias
// MODE 1: bf16 out = silu(acc+bias)
// MODE 2: bf16 out = acc+bias
// MODE 3: f32 out = alpha*(acc+bias) + res
template <int MODE>
__global__ __launch_bounds__(256) void gemm_kernel(const bf16* __restrict__ A,
                                                   const bf16* __restrict__ Bt,
                                                   const float* __restrict__ bias,
                                                   const float* __restrict__ res,
                                                   float alpha,
                                                   float* __restrict__ outF,
                                                   bf16* __restrict__ outB,
                                                   int M, int N, int Kd) {
  __shared__ __align__(16) bf16 As[2][128][72];
  __shared__ __align__(16) bf16 Bs[2][128][72];
  const int tid = threadIdx.x;
  const int lane = tid & 31, w = tid >> 5;
  const int lane15 = lane & 15, hi = lane >> 4, kb = hi * 8;
  const int m0 = blockIdx.y * 128, n0 = blockIdx.x * 128;
  const int wm = (w & 1) * 64, wn = (w >> 1) * 32;

  v8f acc[4][2];
#pragma unroll
  for (int s = 0; s < 4; ++s)
#pragma unroll
    for (int u = 0; u < 2; ++u) {
      v8f z = {0.f, 0.f, 0.f, 0.f, 0.f, 0.f, 0.f, 0.f};
      acc[s][u] = z;
    }

  // 128 rows x 64 cols bf16 per tile = 1024 x b128 -> 4 per thread per matrix
  auto issue = [&](int p, int kt) {
#pragma unroll
    for (int it = 0; it < 4; ++it) {
      int tau = tid + it * 256;
      int r = tau >> 3, sg = (tau & 7) * 8;
      ASYNC_LDS_B128(&A[(size_t)(m0 + r) * Kd + kt + sg], &As[p][r][sg]);
      ASYNC_LDS_B128(&Bt[(size_t)(n0 + r) * Kd + kt + sg], &Bs[p][r][sg]);
    }
  };
  auto compute = [&](int p) {
#pragma unroll
    for (int c = 0; c < 2; ++c) {
      v16bf af[4], bfm[2];
#pragma unroll
      for (int s = 0; s < 4; ++s) {
        const bf16* rp = &As[p][wm + s * 16 + lane15][c * 32 + kb];
        af[s] = make_frag(*(const bf16x8*)rp, *(const bf16x8*)(rp + 16));
      }
#pragma unroll
      for (int u = 0; u < 2; ++u) {
        const bf16* rp = &Bs[p][wn + u * 16 + lane15][c * 32 + kb];
        bfm[u] = make_frag(*(const bf16x8*)rp, *(const bf16x8*)(rp + 16));
      }
#pragma unroll
      for (int s = 0; s < 4; ++s)
#pragma unroll
        for (int u = 0; u < 2; ++u)
          acc[s][u] = __builtin_amdgcn_wmma_f32_16x16x32_bf16(
              false, af[s], false, bfm[u], (short)0, acc[s][u], false, false);
    }
  };

  issue(0, 0);
  int p = 0;
  int kt = 0;
  for (; kt + 64 < Kd; kt += 64) {
    __syncthreads();                       // close reads of buf[p^1]
    issue(p ^ 1, kt + 64);                 // prefetch next tile (DMA)
    __builtin_amdgcn_s_wait_asynccnt(8);   // current tile's 8 DMAs done (in-order)
    __syncthreads();                       // visible to all waves
    compute(p);
    p ^= 1;
  }
  __syncthreads();
  __builtin_amdgcn_s_wait_asynccnt(0);
  __syncthreads();
  compute(p);

  // epilogue
#pragma unroll
  for (int u = 0; u < 2; ++u) {
    const int ng = n0 + wn + u * 16 + lane15;
    const float bv = bias[ng];
#pragma unroll
    for (int s = 0; s < 4; ++s) {
#pragma unroll
      for (int i = 0; i < 8; ++i) {
        int mg = m0 + wm + s * 16 + i + 8 * hi;
        float v = acc[s][u][i] + bv;
        size_t off = (size_t)mg * N + ng;
        if (MODE == 0)      outF[off] = v;
        else if (MODE == 1) outB[off] = f2b(v * sigmoidf_(v));
        else if (MODE == 2) outB[off] = f2b(v);
        else                outF[off] = alpha * v + res[off];
      }
    }
  }
}

// ---------------- RoPE + head pack ----------------
// qkv[(t*B+b),1536] -> Q/K [bh][t][hd] ; V transposed [bh][hd][t]
__global__ __launch_bounds__(256) void rope_pack_kernel(const bf16* __restrict__ qkv,
                                                        bf16* __restrict__ Qp,
                                                        bf16* __restrict__ Kp,
                                                        bf16* __restrict__ VpT) {
  int idx = blockIdx.x * 256 + threadIdx.x;   // [0, B*H*T*HD)
  int hd = idx & 63;
  int t = (idx >> 6) & 1023;
  int h = (idx >> 16) & 7;
  int b = idx >> 19;
  size_t base = (size_t)(t * Bb + b) * (3 * Dm) + h * HDd;
  float qv = b2f(qkv[base + hd]);
  float kv = b2f(qkv[base + Dm + hd]);
  float vv = b2f(qkv[base + 2 * Dm + hd]);
  int hd2 = (hd < 32) ? hd + 32 : hd - 32;
  float qo = b2f(qkv[base + hd2]);
  float ko = b2f(qkv[base + Dm + hd2]);
  float sgn = (hd < 32) ? -1.f : 1.f;
  int i = hd & 31;
  float inv = __expf(-(float)i * 0.28782313662425575f);   // ln(10000)/32
  float th = (float)t * inv;
  float c = __cosf(th), s = __sinf(th);
  int bh = b * Hh + h;
  size_t o = (size_t)idx;                       // ((b*H+h)*T + t)*HD + hd
  Qp[o] = f2b(qv * c + sgn * qo * s);
  Kp[o] = f2b(kv * c + sgn * ko * s);
  VpT[((size_t)bh * HDd + hd) * Tt + t] = f2b(vv);
}

// ---------------- fused flash attention (64-wide KV chunks) ----------------
__global__ __launch_bounds__(256) void attn_kernel(const bf16* __restrict__ Qp,
                                                   const bf16* __restrict__ Kp,
                                                   const bf16* __restrict__ VpT,
                                                   const unsigned char* __restrict__ kpm,
                                                   bf16* __restrict__ Obf) {
  const int bh = blockIdx.y;               // 0..127
  const int b = bh >> 3, h = bh & 7;
  const int tid = threadIdx.x, w = tid >> 5, lane = tid & 31;
  const int lane15 = lane & 15, hi = lane >> 4, kb = hi * 8;
  const int q0 = blockIdx.x * 128 + w * 16;

  __shared__ __align__(16) bf16 Ks[2][64][72];   // [kv][hd]
  __shared__ __align__(16) bf16 Vt[2][64][72];   // [hd][kv]
  __shared__ __align__(16) bf16 Ps[8][16][72];   // wave-private P staging (16 m x 64 kv)

  const size_t bhbase = (size_t)bh * Tt * HDd;
  const size_t vbase  = (size_t)bh * HDd * Tt;

  // Q fragments (16 x 64 split into two K=32 chunks)
  v16bf qf[2];
  {
    const bf16* qrow = Qp + bhbase + (size_t)(q0 + lane15) * HDd;
#pragma unroll
    for (int c = 0; c < 2; ++c)
      qf[c] = make_frag(*(const bf16x8*)&qrow[c * 32 + kb],
                        *(const bf16x8*)&qrow[c * 32 + 16 + kb]);
  }

  v8f oacc[4];
#pragma unroll
  for (int n = 0; n < 4; ++n) {
    v8f z = {0.f, 0.f, 0.f, 0.f, 0.f, 0.f, 0.f, 0.f};
    oacc[n] = z;
  }
  float mrun[8], lpart[8];
#pragma unroll
  for (int i = 0; i < 8; ++i) { mrun[i] = -1e30f; lpart[i] = 0.f; }

  // K tile: 64 x 64 -> 512 b128 (2/thread); V^T tile: 64 x 64 -> 512 b128 (2/thread)
  auto issueKV = [&](int p, int kv0) {
#pragma unroll
    for (int it = 0; it < 2; ++it) {
      int tau = tid + it * 256;
      int r = tau >> 3, sg = (tau & 7) * 8;
      ASYNC_LDS_B128(&Kp[bhbase + (size_t)(kv0 + r) * HDd + sg], &Ks[p][r][sg]);
      ASYNC_LDS_B128(&VpT[vbase + (size_t)r * Tt + kv0 + sg], &Vt[p][r][sg]);
    }
  };

  auto compute_chunk = [&](int p, int kv0) {
    // S = Q * K^T : four 16x16 n-tiles, each K-dim 64 = 2 WMMAs
    v8f sac[4];
#pragma unroll
    for (int u = 0; u < 4; ++u) {
      v8f z = {0.f, 0.f, 0.f, 0.f, 0.f, 0.f, 0.f, 0.f};
      sac[u] = z;
      const bf16* krow = &Ks[p][u * 16 + lane15][0];
#pragma unroll
      for (int c = 0; c < 2; ++c) {
        v16bf kf = make_frag(*(const bf16x8*)&krow[c * 32 + kb],
                             *(const bf16x8*)&krow[c * 32 + 16 + kb]);
        sac[u] = __builtin_amdgcn_wmma_f32_16x16x32_bf16(
            false, qf[c], false, kf, (short)0, sac[u], false, false);
      }
    }
    float mb[4];
#pragma unroll
    for (int u = 0; u < 4; ++u)
      mb[u] = kpm[b * Tt + kv0 + u * 16 + lane15] ? -1e9f : 0.f;

    // online softmax; only the row-max needs a cross-lane reduction.
    // lpart stays lane-private (reduced once at the end, by linearity).
#pragma unroll
    for (int i = 0; i < 8; ++i) {
      float v0 = sac[0][i] * 0.125f + mb[0];
      float v1 = sac[1][i] * 0.125f + mb[1];
      float v2 = sac[2][i] * 0.125f + mb[2];
      float v3 = sac[3][i] * 0.125f + mb[3];
      float mx = redmax16(fmaxf(fmaxf(v0, v1), fmaxf(v2, v3)));
      float mN = fmaxf(mrun[i], mx);
      float corr = __expf(mrun[i] - mN);
      float p0 = __expf(v0 - mN);
      float p1 = __expf(v1 - mN);
      float p2 = __expf(v2 - mN);
      float p3 = __expf(v3 - mN);
      lpart[i] = lpart[i] * corr + (p0 + p1) + (p2 + p3);
      mrun[i] = mN;
#pragma unroll
      for (int n = 0; n < 4; ++n) oacc[n][i] *= corr;
      bf16* prow = &Ps[w][i + 8 * hi][0];
      prow[lane15]      = f2b(p0);
      prow[16 + lane15] = f2b(p1);
      prow[32 + lane15] = f2b(p2);
      prow[48 + lane15] = f2b(p3);
    }

    // O += P * V : P is 16x64 (2 A-frags), V^T gives 4 n-tiles x 2 k-chunks
    v16bf pf[2];
    {
      const bf16* prow = &Ps[w][lane15][0];
#pragma unroll
      for (int cc = 0; cc < 2; ++cc)
        pf[cc] = make_frag(*(const bf16x8*)&prow[cc * 32 + kb],
                           *(const bf16x8*)&prow[cc * 32 + 16 + kb]);
    }
#pragma unroll
    for (int n = 0; n < 4; ++n) {
      const bf16* vrow = &Vt[p][n * 16 + lane15][0];
#pragma unroll
      for (int cc = 0; cc < 2; ++cc) {
        v16bf vf = make_frag(*(const bf16x8*)&vrow[cc * 32 + kb],
                             *(const bf16x8*)&vrow[cc * 32 + 16 + kb]);
        oacc[n] = __builtin_amdgcn_wmma_f32_16x16x32_bf16(
            false, pf[cc], false, vf, (short)0, oacc[n], false, false);
      }
    }
  };

  issueKV(0, 0);
  int p = 0;
  for (int kc = 0; kc < Tt / 64 - 1; ++kc) {
    __syncthreads();                       // close reads of buf[p^1]
    issueKV(p ^ 1, (kc + 1) * 64);         // prefetch next KV chunk
    __builtin_amdgcn_s_wait_asynccnt(4);   // current chunk's 4 DMAs done
    __syncthreads();
    compute_chunk(p, kc * 64);
    p ^= 1;
  }
  __syncthreads();
  __builtin_amdgcn_s_wait_asynccnt(0);
  __syncthreads();
  compute_chunk(p, Tt - 64);

  // write O rows in (t*B+b, h*64+hd) layout (gemm-ready)
#pragma unroll
  for (int i = 0; i < 8; ++i) {
    float inv = 1.f / redsum16(lpart[i]);
    int t = q0 + i + 8 * hi;
    size_t rb = ((size_t)t * Bb + b) * Dm + h * HDd;
#pragma unroll
    for (int n = 0; n < 4; ++n)
      Obf[rb + n * 16 + lane15] = f2b(oacc[n][i] * inv);
  }
}

// ---------------- GLU over channels ----------------
__global__ __launch_bounds__(256) void glu_kernel(const float* __restrict__ pw1,
                                                  float* __restrict__ hglu) {
  int idx = blockIdx.x * 256 + threadIdx.x;     // [0, NROW*512)
  int r = idx >> 9, c = idx & 511;
  float a = pw1[(size_t)r * 1024 + c];
  float g = pw1[(size_t)r * 1024 + 512 + c];
  hglu[idx] = a * sigmoidf_(g);
}

// ---------------- depthwise conv K=31, LDS-tiled ----------------
// grid: (t-tiles=32, c-tiles=8, b=16); block tile = 32 t x 64 c (+30 t halo)
__global__ __launch_bounds__(256) void dwconv_kernel(const float* __restrict__ hglu,
                                                     const float* __restrict__ wdw,
                                                     const float* __restrict__ bdw,
                                                     float* __restrict__ dwout) {
  __shared__ __align__(16) float tile[62][64];
  const int b = blockIdx.z, c0 = blockIdx.y * 64, t0 = blockIdx.x * 32;
  for (int task = threadIdx.x; task < 62 * 16; task += 256) {
    int rr = task >> 4, sg = (task & 15) * 4;
    int ts = t0 - 15 + rr;
    float4 v = make_float4(0.f, 0.f, 0.f, 0.f);
    if (ts >= 0 && ts < Tt)
      v = *(const float4*)&hglu[((size_t)(ts * Bb + b) << 9) + c0 + sg];
    *(float4*)&tile[rr][sg] = v;
  }
  __syncthreads();
  const int cl = threadIdx.x & 63, tl = threadIdx.x >> 6;
  float wv[KK];
#pragma unroll
  for (int k = 0; k < KK; ++k) wv[k] = wdw[(c0 + cl) * KK + k];
  const float bb = bdw[c0 + cl];
#pragma unroll
  for (int j = 0; j < 8; ++j) {
    int t = tl + j * 4;
    float acc = bb;
#pragma unroll
    for (int k = 0; k < KK; ++k) acc += tile[t + k][cl] * wv[k];
    dwout[((size_t)((t0 + t) * Bb + b) << 9) + c0 + cl] = acc;
  }
}

// ---------------- GroupNorm(1 group) stats per sample ----------------
__global__ __launch_bounds__(256) void gn_stats_kernel(const float* __restrict__ dw,
                                                       float* __restrict__ stats) {
  const int b = blockIdx.x;
  float s = 0.f, q = 0.f;
  for (int e = threadIdx.x; e < Dm * Tt; e += 256) {
    float v = dw[((size_t)(e >> 9)) * (Bb * Dm) + b * Dm + (e & 511)];
    s += v; q += v * v;
  }
  __shared__ float ss[256], qq[256];
  ss[threadIdx.x] = s; qq[threadIdx.x] = q;
  __syncthreads();
  for (int st = 128; st > 0; st >>= 1) {
    if (threadIdx.x < st) { ss[threadIdx.x] += ss[threadIdx.x + st]; qq[threadIdx.x] += qq[threadIdx.x + st]; }
    __syncthreads();
  }
  if (threadIdx.x == 0) {
    const float inv = 1.f / (float)(Dm * Tt);
    float mean = ss[0] * inv;
    float var = qq[0] * inv - mean * mean;
    stats[2 * b] = mean;
    stats[2 * b + 1] = rsqrtf(var + EPSf);
  }
}

__global__ __launch_bounds__(256) void gn_silu_kernel(const float* __restrict__ dw,
                                                      const float* __restrict__ stats,
                                                      const float* __restrict__ gg,
                                                      const float* __restrict__ gb,
                                                      bf16* __restrict__ out) {
  int idx = blockIdx.x * 256 + threadIdx.x;
  int b = (idx >> 9) & 15;
  int c = idx & 511;
  float v = (dw[idx] - stats[2 * b]) * stats[2 * b + 1] * gg[c] + gb[c];
  out[idx] = f2b(v * sigmoidf_(v));
}

// ---------------- host launch ----------------
extern "C" void kernel_launch(void* const* d_in, const int* in_sizes, int n_in,
                              void* d_out, int out_size, void* d_ws, size_t ws_size,
                              hipStream_t stream) {
  (void)in_sizes; (void)n_in; (void)out_size; (void)ws_size;
  const float* x0      = (const float*)d_in[0];
  const unsigned char* kpm = (const unsigned char*)d_in[1];
  const float* ffn1_g  = (const float*)d_in[2];
  const float* ffn1_b  = (const float*)d_in[3];
  const float* ffn1_w1 = (const float*)d_in[4];
  const float* ffn1_b1 = (const float*)d_in[5];
  const float* ffn1_w2 = (const float*)d_in[6];
  const float* ffn1_b2 = (const float*)d_in[7];
  const float* attn_g  = (const float*)d_in[8];
  const float* attn_b  = (const float*)d_in[9];
  const float* qkv_w   = (const float*)d_in[10];
  const float* qkv_b   = (const float*)d_in[11];
  const float* out_w   = (const float*)d_in[12];
  const float* out_b   = (const float*)d_in[13];
  const float* cln_g   = (const float*)d_in[14];
  const float* cln_b   = (const float*)d_in[15];
  const float* pw1_w   = (const float*)d_in[16];
  const float* pw1_b   = (const float*)d_in[17];
  const float* dw_w    = (const float*)d_in[18];
  const float* dw_b    = (const float*)d_in[19];
  const float* gn_g    = (const float*)d_in[20];
  const float* gn_b    = (const float*)d_in[21];
  const float* pw2_w   = (const float*)d_in[22];
  const float* pw2_b   = (const float*)d_in[23];
  const float* ffn2_g  = (const float*)d_in[24];
  const float* ffn2_b  = (const float*)d_in[25];
  const float* ffn2_w1 = (const float*)d_in[26];
  const float* ffn2_b1 = (const float*)d_in[27];
  const float* ffn2_w2 = (const float*)d_in[28];
  const float* ffn2_b2 = (const float*)d_in[29];
  const float* fin_g   = (const float*)d_in[30];
  const float* fin_b   = (const float*)d_in[31];

  char* ws = (char*)d_ws;
  size_t off = 0;
  auto alloc = [&](size_t bytes) { size_t r = off; off += (bytes + 255) & ~(size_t)255; return (void*)(ws + r); };

  // transposed bf16 weights [N][K]
  bf16* bw_f1w1 = (bf16*)alloc((size_t)Dm * Fm * 2);
  bf16* bw_f1w2 = (bf16*)alloc((size_t)Fm * Dm * 2);
  bf16* bw_qkv  = (bf16*)alloc((size_t)Dm * 3 * Dm * 2);
  bf16* bw_out  = (bf16*)alloc((size_t)Dm * Dm * 2);
  bf16* bw_pw1  = (bf16*)alloc((size_t)Dm * 2 * Dm * 2);
  bf16* bw_pw2  = (bf16*)alloc((size_t)Dm * Dm * 2);
  bf16* bw_f2w1 = (bf16*)alloc((size_t)Dm * Fm * 2);
  bf16* bw_f2w2 = (bf16*)alloc((size_t)Fm * Dm * 2);
  bf16* Abf = (bf16*)alloc((size_t)NROW * Dm * 2);        // LN out / attn-out / conv act
  bf16* Hbf = (bf16*)alloc((size_t)NROW * Fm * 2);        // FFN hidden / qkv_bf / hglu(f32)
  bf16* Qp  = (bf16*)alloc((size_t)Bb * Hh * Tt * HDd * 2);
  bf16* Kp  = (bf16*)alloc((size_t)Bb * Hh * Tt * HDd * 2);
  bf16* VpT = (bf16*)alloc((size_t)Bb * Hh * Tt * HDd * 2);
  float* X1 = (float*)alloc((size_t)NROW * Dm * 4);
  float* X2 = (float*)alloc((size_t)NROW * Dm * 4);
  float* F3 = (float*)alloc((size_t)NROW * 2 * Dm * 4);   // pw1 out / dw out
  float* stats = (float*)alloc((size_t)Bb * 2 * 4);

  const dim3 blk(256);
  auto wtr = [&](const float* s, bf16* d, int Kd, int N) {
    wtr_kernel<<<dim3(N / 32, Kd / 32), blk, 0, stream>>>(s, d, Kd, N);
  };
  wtr(ffn1_w1, bw_f1w1, Dm, Fm);
  wtr(ffn1_w2, bw_f1w2, Fm, Dm);
  wtr(qkv_w,   bw_qkv,  Dm, 3 * Dm);
  wtr(out_w,   bw_out,  Dm, Dm);
  wtr(pw1_w,   bw_pw1,  Dm, 2 * Dm);
  wtr(pw2_w,   bw_pw2,  Dm, Dm);
  wtr(ffn2_w1, bw_f2w1, Dm, Fm);
  wtr(ffn2_w2, bw_f2w2, Fm, Dm);

  const dim3 lng(NROW / 8);
  const int elgrid = NROW * Dm / 256;   // 32768

  // ---- FFN1 ----
  ln_kernel<0><<<lng, blk, 0, stream>>>(x0, ffn1_g, ffn1_b, Abf, nullptr);
  gemm_kernel<1><<<dim3(Fm / 128, NROW / 128), blk, 0, stream>>>(
      Abf, bw_f1w1, ffn1_b1, nullptr, 0.f, nullptr, Hbf, NROW, Fm, Dm);
  gemm_kernel<3><<<dim3(Dm / 128, NROW / 128), blk, 0, stream>>>(
      Hbf, bw_f1w2, ffn1_b2, x0, 0.5f, X1, nullptr, NROW, Dm, Fm);

  // ---- Attention ----
  ln_kernel<0><<<lng, blk, 0, stream>>>(X1, attn_g, attn_b, Abf, nullptr);
  gemm_kernel<2><<<dim3(3 * Dm / 128, NROW / 128), blk, 0, stream>>>(
      Abf, bw_qkv, qkv_b, nullptr, 0.f, nullptr, Hbf, NROW, 3 * Dm, Dm);
  rope_pack_kernel<<<Bb * Hh * Tt * HDd / 256, blk, 0, stream>>>(Hbf, Qp, Kp, VpT);
  attn_kernel<<<dim3(Tt / 128, Bb * Hh), blk, 0, stream>>>(Qp, Kp, VpT, kpm, Abf);
  gemm_kernel<3><<<dim3(Dm / 128, NROW / 128), blk, 0, stream>>>(
      Abf, bw_out, out_b, X1, 1.f, X2, nullptr, NROW, Dm, Dm);

  // ---- Conv module ----
  ln_kernel<0><<<lng, blk, 0, stream>>>(X2, cln_g, cln_b, Abf, nullptr);
  gemm_kernel<0><<<dim3(2 * Dm / 128, NROW / 128), blk, 0, stream>>>(
      Abf, bw_pw1, pw1_b, nullptr, 0.f, F3, nullptr, NROW, 2 * Dm, Dm);
  glu_kernel<<<elgrid, blk, 0, stream>>>(F3, (float*)Hbf);
  dwconv_kernel<<<dim3(Tt / 32, Dm / 64, Bb), blk, 0, stream>>>((float*)Hbf, dw_w, dw_b, F3);
  gn_stats_kernel<<<Bb, blk, 0, stream>>>(F3, stats);
  gn_silu_kernel<<<elgrid, blk, 0, stream>>>(F3, stats, gn_g, gn_b, Abf);
  gemm_kernel<3><<<dim3(Dm / 128, NROW / 128), blk, 0, stream>>>(
      Abf, bw_pw2, pw2_b, X2, 1.f, X1, nullptr, NROW, Dm, Dm);

  // ---- FFN2 ----
  ln_kernel<0><<<lng, blk, 0, stream>>>(X1, ffn2_g, ffn2_b, Abf, nullptr);
  gemm_kernel<1><<<dim3(Fm / 128, NROW / 128), blk, 0, stream>>>(
      Abf, bw_f2w1, ffn2_b1, nullptr, 0.f, nullptr, Hbf, NROW, Fm, Dm);
  gemm_kernel<3><<<dim3(Dm / 128, NROW / 128), blk, 0, stream>>>(
      Hbf, bw_f2w2, ffn2_b2, X1, 0.5f, X2, nullptr, NROW, Dm, Fm);

  // ---- Final LN ----
  ln_kernel<1><<<lng, blk, 0, stream>>>(X2, fin_g, fin_b, nullptr, (float*)d_out);
}
